// MultiHeadCrossAttention1_55937654063674
// MI455X (gfx1250) — matrix operand order repaired and checked by reading, not compiled, not run on hardware
//
#include <hip/hip_runtime.h>
#include <hip/hip_bf16.h>

typedef __bf16 bf16;
typedef __attribute__((ext_vector_type(16))) __bf16 v16bf;
typedef __attribute__((ext_vector_type(8)))  float  v8f;

union FragBF { v16bf v; uint4 q[2]; };
union FragF  { v8f   v; float f[8]; };

__device__ __forceinline__ v8f wmma_bf16(v16bf a, v16bf b, v8f c) {
  return __builtin_amdgcn_wmma_f32_16x16x32_bf16(false, a, false, b, (short)0, c, false, false);
}

// ---------------------------------------------------------------- prep kernels

__global__ void gate_kernel(const float* __restrict__ glog, float* __restrict__ gate) {
  if (threadIdx.x == 0) {
    float g = 1.0f;
    for (int i = 0; i < 30; ++i) {
      float s = 1.0f / (1.0f + __expf(-glog[i]));   // sigmoid
      g *= (1.0f - s);                              // cumprod(1 - sigmoid)
      gate[i] = g;
    }
    gate[30] = 1.0f; gate[31] = 1.0f;
  }
}

// Transposed bf16 weights: wkvT[j][k], j<1024 -> Wk[k][j], j>=1024 -> Wv[k][j-1024]
__global__ void conv_wkv_kernel(const float* __restrict__ Wk, const float* __restrict__ Wv,
                                bf16* __restrict__ outT) {
  unsigned idx = blockIdx.x * 256u + threadIdx.x;   // k*2048 + j (j fastest: coalesced reads)
  unsigned k = idx >> 11, j = idx & 2047u;
  float v = (j < 1024u) ? Wk[k * 1024u + j] : Wv[k * 1024u + (j - 1024u)];
  outT[j * 1024u + k] = (bf16)v;                    // transposed store (one-shot kernel)
}

// wfcT[n][k] = Wfc[k][n]  (bf16)
__global__ void conv_wfc_kernel(const float* __restrict__ Wfc, bf16* __restrict__ outT) {
  unsigned idx = blockIdx.x * 256u + threadIdx.x;   // k*1024 + n
  unsigned k = idx >> 10, n = idx & 1023u;
  outT[n * 1024u + k] = (bf16)Wfc[idx];
}

// q = (q_tok @ Wq) / sqrt(64), stored [h][n][dk] bf16
__global__ void qproj_kernel(const float* __restrict__ qtok, const float* __restrict__ Wq,
                             bf16* __restrict__ qbf) {
  unsigned idx = blockIdx.x * 256u + threadIdx.x;   // n*1024 + c
  unsigned n = idx >> 10, c = idx & 1023u;
  float acc = 0.0f;
  for (unsigned d = 0; d < 1024u; ++d)
    acc += qtok[n * 1024u + d] * Wq[d * 1024u + c];
  unsigned h = c >> 6, dk = c & 63u;
  qbf[h * 4096u + n * 64u + dk] = (bf16)(acc * 0.125f);
}

// ---------------------------------------------------------------- WMMA GEMM
// C[M x Nn] = f32(A[M x K]) @ BT[Nn x K](bf16, pre-transposed), f32 accumulate.
// Double-buffered LDS pipeline; grid = (N-tiles, M-tiles) so the 32 blocks that
// share one A slab are consecutive -> A slab stays L2-resident (x >> L2 size).
// mode 0: out bf16; gate rows l<30 on columns >=1024 (V half).
// mode 1: out f32 with residual resid[(m&63)*1024 + n].
__global__ __launch_bounds__(256) void gemm_kernel(
    const float* __restrict__ A, const bf16* __restrict__ BT,
    bf16* __restrict__ outBF, float* __restrict__ outF,
    const float* __restrict__ gate, const float* __restrict__ resid,
    int M, int Nn, int K, int mode)
{
  __shared__ __align__(16) bf16 sA[2][128 * 40];  // [m][k] pad->40 (80B rows, 16B aligned)
  __shared__ __align__(16) bf16 sB[2][64 * 40];   // [n][k]

  const int tid  = threadIdx.x;
  const int lane = tid & 31, wid = tid >> 5;
  const int mw = (wid >> 1) * 32, nw = (wid & 1) * 32;
  const int lm  = lane & 15;
  const int kbA = (lane >> 4) * 8;              // A frag K base (per ISA A-layout)
  const int kbB = (lane >> 4) * 8;              // B frag K base (same layout as A)
  const int n0 = blockIdx.x * 64;               // N fastest: A-slab reuse hits L2
  const int m0 = blockIdx.y * 128;

  // per-thread staging coords (32-bit addressing throughout)
  const unsigned aRow = (unsigned)(tid >> 3), aC4 = (unsigned)((tid & 7) * 4);
  const unsigned bN = (unsigned)(tid >> 2), bKq = (unsigned)((tid & 3) * 8);
  const unsigned aBase = (unsigned)(m0 + aRow) * (unsigned)K + aC4;   // + 32*ii*K later
  const unsigned bBase = (unsigned)(n0 + bN) * (unsigned)K + bKq;

  FragF acc[2][2];
  for (int i = 0; i < 2; ++i) for (int j = 0; j < 2; ++j)
    for (int e = 0; e < 8; ++e) acc[i][j].f[e] = 0.0f;

  // register-staged tiles (so global loads clause & overlap with WMMA)
  float4 av[4]; uint4 bv;
  #pragma unroll
  for (unsigned ii = 0; ii < 4; ++ii)
    av[ii] = *(const float4*)&A[aBase + 32u * ii * (unsigned)K];
  bv = *(const uint4*)&BT[bBase];

  int cur = 0;
  for (int k0 = 0; k0 < K; k0 += 32) {
    // commit staged registers to LDS buffer `cur`
    #pragma unroll
    for (unsigned ii = 0; ii < 4; ++ii) {
      bf16* d = &sA[cur][(aRow + 32u * ii) * 40u + aC4];
      d[0] = (bf16)av[ii].x; d[1] = (bf16)av[ii].y;
      d[2] = (bf16)av[ii].z; d[3] = (bf16)av[ii].w;
    }
    *(uint4*)&sB[cur][bN * 40u + bKq] = bv;
    __syncthreads();

    // issue next slab's global loads (overlap with WMMA below)
    const int k1 = k0 + 32;
    if (k1 < K) {
      #pragma unroll
      for (unsigned ii = 0; ii < 4; ++ii)
        av[ii] = *(const float4*)&A[aBase + (unsigned)k1 + 32u * ii * (unsigned)K];
      bv = *(const uint4*)&BT[bBase + (unsigned)k1];
      if (k1 + 32 < K) {  // gfx1250 global_prefetch_b8 two slabs ahead
        __builtin_prefetch(&A[aBase + (unsigned)(k1 + 32)], 0, 1);
        __builtin_prefetch(&BT[bBase + (unsigned)(k1 + 32)], 0, 1);
      }
    }

    FragBF a0, a1, b0, b1;
    { const bf16* p = &sA[cur][(mw      + lm) * 40 + kbA]; a0.q[0] = *(const uint4*)p; a0.q[1] = *(const uint4*)(p + 16); }
    { const bf16* p = &sA[cur][(mw + 16 + lm) * 40 + kbA]; a1.q[0] = *(const uint4*)p; a1.q[1] = *(const uint4*)(p + 16); }
    { const bf16* p = &sB[cur][(nw      + lm) * 40 + kbB]; b0.q[0] = *(const uint4*)p; b0.q[1] = *(const uint4*)(p + 16); }
    { const bf16* p = &sB[cur][(nw + 16 + lm) * 40 + kbB]; b1.q[0] = *(const uint4*)p; b1.q[1] = *(const uint4*)(p + 16); }

    acc[0][0].v = wmma_bf16(a0.v, b0.v, acc[0][0].v);
    acc[0][1].v = wmma_bf16(a0.v, b1.v, acc[0][1].v);
    acc[1][0].v = wmma_bf16(a1.v, b0.v, acc[1][0].v);
    acc[1][1].v = wmma_bf16(a1.v, b1.v, acc[1][1].v);
    cur ^= 1;   // double buffer: next store goes to other half, one barrier/iter
  }

  const int hi8 = (lane >> 4) * 8;
  for (int i = 0; i < 2; ++i) for (int j = 0; j < 2; ++j) {
    unsigned gn  = (unsigned)(n0 + nw + j * 16 + lm);
    unsigned gmb = (unsigned)(m0 + mw + i * 16 + hi8);
    for (unsigned e = 0; e < 8; ++e) {
      unsigned gm = gmb + e;
      float v = acc[i][j].f[e];
      if (mode == 0) {
        unsigned l = gm & 2047u;                  // row = b*2048 + l
        if (gn >= 1024u && l < 30u) v *= gate[l]; // gate*(x@Wv) == (gate*x)@Wv
        outBF[gm * (unsigned)Nn + gn] = (bf16)v;
      } else {
        outF[gm * (unsigned)Nn + gn] = v + resid[(gm & 63u) * 1024u + gn];
      }
    }
  }
}

// ---------------------------------------------------------------- flash attention
// One block per (b,h). 4 waves; wave w owns query rows [16w,16w+16), streams L=2048
// in chunks of 32 with online softmax. kv layout: [b*2048+l][j], j<1024 = K, else V.
__global__ __launch_bounds__(128) void attn_kernel(const bf16* __restrict__ qbf,
                                                   const bf16* __restrict__ kv,
                                                   float* __restrict__ aout)
{
  __shared__ __align__(16) bf16  sQ[64 * 72];      // [n][d]
  __shared__ __align__(16) bf16  sK[32 * 72];      // [l][d]
  __shared__ __align__(16) bf16  sVt[64 * 40];     // [dv][l]
  __shared__ __align__(16) float sS[4][16][32];    // per-wave scores
  __shared__ __align__(16) bf16  sP[4][16][40];    // per-wave probs (A-frag layout rows)
  __shared__ float sScale[4][16];
  __shared__ float sLsum[4][16];

  const unsigned b = blockIdx.x >> 4, h = blockIdx.x & 15u;
  const int tid = threadIdx.x;
  const int lane = tid & 31, w = tid >> 5;
  const int lm  = lane & 15;
  const int kbA = (lane >> 4) * 8;
  const int kbB = (lane >> 4) * 8;
  const int hi8 = (lane >> 4) * 8;

  // stage q tile for this head: 512 uint4 / 128 threads = 4 each
  for (unsigned ii = 0; ii < 4; ++ii) {
    unsigned e = (unsigned)tid + 128u * ii;        // uint4 slot: n = e>>3, dq = (e&7)*8
    unsigned n = e >> 3, dq = (e & 7u) * 8u;
    *(uint4*)&sQ[n * 72u + dq] = *(const uint4*)&qbf[h * 4096u + n * 64u + dq];
  }
  __syncthreads();

  // per-wave q A-frags (built once, reused for all 64 chunks)
  FragBF qa[2];
  for (int dh = 0; dh < 2; ++dh) {
    const bf16* p = &sQ[(w * 16 + lm) * 72 + dh * 32 + kbA];
    qa[dh].q[0] = *(const uint4*)p; qa[dh].q[1] = *(const uint4*)(p + 16);
  }

  FragF oacc[4];
  for (int t = 0; t < 4; ++t) for (int e = 0; e < 8; ++e) oacc[t].f[e] = 0.0f;
  float m_r = -1e30f, l_r = 0.0f;                  // valid in lanes 0..15

  const unsigned kvBase = (b * 2048u) * 2048u + h * 64u;
  const unsigned sLL = (unsigned)tid >> 3;          // staging row (per uint4 slot)
  const unsigned sDq = ((unsigned)tid & 7u) * 8u;

  for (unsigned l0 = 0; l0 < 2048u; l0 += 32u) {
    // stage K [32][64] and V transposed [64][32]; batch loads, then stores
    uint4 kreg[2], vreg[2];
    #pragma unroll
    for (unsigned ii = 0; ii < 2; ++ii) {
      unsigned row = kvBase + (l0 + sLL + 16u * ii) * 2048u;
      kreg[ii] = *(const uint4*)&kv[row + sDq];
      vreg[ii] = *(const uint4*)&kv[row + 1024u + sDq];
    }
    #pragma unroll
    for (unsigned ii = 0; ii < 2; ++ii) {
      *(uint4*)&sK[(sLL + 16u * ii) * 72u + sDq] = kreg[ii];
      const bf16* pv = (const bf16*)&vreg[ii];
      #pragma unroll
      for (unsigned t = 0; t < 8; ++t) sVt[(sDq + t) * 40u + sLL + 16u * ii] = pv[t];
    }
    __syncthreads();

    // S = q @ K^T   (two 16-l subtiles, d split 2x32)
    for (int ls = 0; ls < 2; ++ls) {
      FragF s; for (int e = 0; e < 8; ++e) s.f[e] = 0.0f;
      for (int dh = 0; dh < 2; ++dh) {
        FragBF kf;
        const bf16* p = &sK[(ls * 16 + lm) * 72 + dh * 32 + kbB];
        kf.q[0] = *(const uint4*)p; kf.q[1] = *(const uint4*)(p + 16);
        s.v = wmma_bf16(qa[dh].v, kf.v, s.v);
      }
      for (int e = 0; e < 8; ++e)
        sS[w][hi8 + e][ls * 16 + lm] = s.f[e];     // wave-local LDS, in-order
    }

    // online softmax: lanes 0..15 each own one query row
    if (lane < 16) {
      int r = lane;
      float cmax = -1e30f;
      for (int c = 0; c < 32; ++c) cmax = fmaxf(cmax, sS[w][r][c]);
      float nm = fmaxf(m_r, cmax);
      float sc = __expf(m_r - nm);
      float sum = 0.0f;
      for (int c = 0; c < 32; ++c) {
        float p = __expf(sS[w][r][c] - nm);
        sP[w][r][c] = (bf16)p;
        sum += p;
      }
      l_r = l_r * sc + sum;
      m_r = nm;
      sScale[w][r] = sc;
    }

    // rescale running O, then O += P @ V
    for (int t = 0; t < 4; ++t)
      for (int e = 0; e < 8; ++e)
        oacc[t].f[e] *= sScale[w][hi8 + e];

    FragBF pf;
    { const bf16* p = &sP[w][lm][kbA]; pf.q[0] = *(const uint4*)p; pf.q[1] = *(const uint4*)(p + 16); }
    for (int t = 0; t < 4; ++t) {
      FragBF vf;
      const bf16* p = &sVt[(t * 16 + lm) * 40 + kbB];
      vf.q[0] = *(const uint4*)p; vf.q[1] = *(const uint4*)(p + 16);
      oacc[t].v = wmma_bf16(pf.v, vf.v, oacc[t].v);
    }
    __syncthreads();
  }

  // finalize: O / l_r, write f32 [b][n][h][dv]
  if (lane < 16) sLsum[w][lane] = l_r;
  for (int t = 0; t < 4; ++t)
    for (int e = 0; e < 8; ++e) {
      int ml = hi8 + e;
      float v = oacc[t].f[e] / sLsum[w][ml];
      unsigned n = (unsigned)(w * 16 + ml);
      aout[(b * 64u + n) * 1024u + h * 64u + (unsigned)(t * 16 + lm)] = v;
    }
}

// ---------------------------------------------------------------- layernorm
__global__ __launch_bounds__(256) void ln_kernel(const float* __restrict__ y,
                                                 const float* __restrict__ gamma,
                                                 const float* __restrict__ beta,
                                                 float* __restrict__ out)
{
  __shared__ float red[256];
  const unsigned r = blockIdx.x, tid = threadIdx.x;
  float s = 0.0f, s2 = 0.0f;
  for (unsigned c = tid; c < 1024u; c += 256u) {
    float v = y[r * 1024u + c];
    s += v; s2 += v * v;
  }
  red[tid] = s; __syncthreads();
  for (int o = 128; o > 0; o >>= 1) { if (tid < (unsigned)o) red[tid] += red[tid + o]; __syncthreads(); }
  float mean = red[0] * (1.0f / 1024.0f);
  __syncthreads();
  red[tid] = s2; __syncthreads();
  for (int o = 128; o > 0; o >>= 1) { if (tid < (unsigned)o) red[tid] += red[tid + o]; __syncthreads(); }
  float var = red[0] * (1.0f / 1024.0f) - mean * mean;
  float inv = rsqrtf(var + 1e-6f);
  for (unsigned c = tid; c < 1024u; c += 256u) {
    float v = y[r * 1024u + c];
    out[r * 1024u + c] = (v - mean) * inv * gamma[c] + beta[c];
  }
}

// ---------------------------------------------------------------- launch

extern "C" void kernel_launch(void* const* d_in, const int* in_sizes, int n_in,
                              void* d_out, int out_size, void* d_ws, size_t ws_size,
                              hipStream_t stream)
{
  const float* x     = (const float*)d_in[0];
  const float* qtok  = (const float*)d_in[1];
  const float* Wq    = (const float*)d_in[2];
  const float* Wk    = (const float*)d_in[3];
  const float* Wv    = (const float*)d_in[4];
  const float* Wfc   = (const float*)d_in[5];
  const float* glog  = (const float*)d_in[6];
  const float* gamma = (const float*)d_in[7];
  const float* beta  = (const float*)d_in[8];

  char* w = (char*)d_ws;
  float* gate = (float*)w;   w += 256;
  bf16*  qbf  = (bf16*)w;    w += (size_t)16 * 64 * 64 * 2;      // 128 KiB
  bf16*  wkvT = (bf16*)w;    w += (size_t)2048 * 1024 * 2;       // 4 MiB   [n][k]
  bf16*  wfcT = (bf16*)w;    w += (size_t)1024 * 1024 * 2;       // 2 MiB   [n][k]
  bf16*  kv   = (bf16*)w;    w += (size_t)65536 * 2048 * 2;      // 256 MiB
  float* aout = (float*)w;   w += (size_t)2048 * 1024 * 4;       // 8 MiB
  float* y    = (float*)w;   /* + 8 MiB */

  gate_kernel<<<1, 32, 0, stream>>>(glog, gate);
  conv_wkv_kernel<<<(1024 * 2048) / 256, 256, 0, stream>>>(Wk, Wv, wkvT);
  conv_wfc_kernel<<<(1024 * 1024) / 256, 256, 0, stream>>>(Wfc, wfcT);
  qproj_kernel<<<256, 256, 0, stream>>>(qtok, Wq, qbf);

  // K|V projection: [65536 x 1024] @ [1024 x 2048]   grid = (N-tiles, M-tiles)
  gemm_kernel<<<dim3(2048 / 64, 65536 / 128), 256, 0, stream>>>(
      x, wkvT, kv, nullptr, gate, nullptr, 65536, 2048, 1024, 0);

  // flash attention per (b,h)
  attn_kernel<<<32 * 16, 128, 0, stream>>>(qbf, kv, aout);

  // FC + residual: [2048 x 1024] @ [1024 x 1024]
  gemm_kernel<<<dim3(1024 / 64, 2048 / 128), 256, 0, stream>>>(
      aout, wfcT, nullptr, y, nullptr, qtok, 2048, 1024, 1024, 1);

  ln_kernel<<<2048, 256, 0, stream>>>(y, gamma, beta, (float*)d_out);
}
